// GraphNet_42838003810849
// MI455X (gfx1250) — matrix operand order repaired and checked by reading, not compiled, not run on hardware
//
#include <hip/hip_runtime.h>
#include <cstddef>

typedef __attribute__((ext_vector_type(16))) _Float16 v16h;
typedef __attribute__((ext_vector_type(8)))  float    v8f;

#define LNEPS 1e-6f

// ---------------------------------------------------------------------------
// Row-wise LayerNorm over an LDS tile. One lane per row (rows <= 16).
// ---------------------------------------------------------------------------
__device__ inline void ln_rows(float* buf, int stride, int nfeat,
                               const float* g, const float* beta, int nrows) {
  const int lane = threadIdx.x & 31;
  if (lane < nrows) {
    float* r = buf + lane * stride;
    float m = 0.f;
    for (int i = 0; i < nfeat; ++i) m += r[i];
    m /= (float)nfeat;
    float v = 0.f;
    for (int i = 0; i < nfeat; ++i) { float d = r[i] - m; v += d * d; }
    v /= (float)nfeat;
    const float inv = rsqrtf(v + LNEPS);
    for (int i = 0; i < nfeat; ++i) r[i] = (r[i] - m) * inv * g[i] + beta[i];
  }
}

// ---------------------------------------------------------------------------
// 16 x K (f32 in LDS, row-major, stride==K) times K x 128 (f16 weights in
// WMMA-fragment-packed order) -> bias (+optional relu) -> LDS out [16][128].
//
// Packed weight layout: for fragment (kt,nt), lane l, half h (h contiguous):
//   Wp[(((kt*8 + nt)*32) + l)*16 + h] = W[(kt*32 + (h>>3)*16 + (l>>4)*8 + (h&7))*128 + nt*16 + (l&15)]
// so each lane's B fragment is one 32-byte contiguous chunk (2x b128 loads).
// ---------------------------------------------------------------------------
__device__ inline void gemm_tile(const float* Alds, int K, const _Float16* Wp,
                                 const float* bias, bool relu, float* outLds) {
  const int lane = threadIdx.x & 31;
  const int rowA = lane & 15;
  const int hiA  = lane >> 4;          // 0 or 1 (half-wave)
  v8f acc[8];
#pragma unroll
  for (int i = 0; i < 8; ++i)
#pragma unroll
    for (int r = 0; r < 8; ++r) acc[i][r] = 0.0f;

  const int ktiles = K >> 5;
  for (int kt = 0; kt < ktiles; ++kt) {
    // A fragment: 4x 16-byte LDS loads (32B aligned), then pack-convert to f16
    const float4* ap4 = (const float4*)(Alds + rowA * K + kt * 32 + hiA * 8);
    const float4 a0 = ap4[0];
    const float4 a1 = ap4[1];
    const float4 a2 = ap4[4];
    const float4 a3 = ap4[5];
    v16h a;
    a[0]  = (_Float16)a0.x; a[1]  = (_Float16)a0.y;
    a[2]  = (_Float16)a0.z; a[3]  = (_Float16)a0.w;
    a[4]  = (_Float16)a1.x; a[5]  = (_Float16)a1.y;
    a[6]  = (_Float16)a1.z; a[7]  = (_Float16)a1.w;
    a[8]  = (_Float16)a2.x; a[9]  = (_Float16)a2.y;
    a[10] = (_Float16)a2.z; a[11] = (_Float16)a2.w;
    a[12] = (_Float16)a3.x; a[13] = (_Float16)a3.y;
    a[14] = (_Float16)a3.z; a[15] = (_Float16)a3.w;

#pragma unroll
    for (int nt = 0; nt < 8; ++nt) {
      const v16h b = *(const v16h*)(Wp + (((size_t)(kt * 8 + nt) * 32) + lane) * 16);
      acc[nt] = __builtin_amdgcn_wmma_f32_16x16x32_f16(
          false, a, false, b, (short)0, acc[nt], false, false);
    }
  }
  // C layout: VGPR r -> row r + 8*hiA, col = nt*16 + (lane&15)
  const int colC = lane & 15;
#pragma unroll
  for (int nt = 0; nt < 8; ++nt) {
    const int col = nt * 16 + colC;
    const float bv = bias[col];
#pragma unroll
    for (int r = 0; r < 8; ++r) {
      const int row = r + hiA * 8;
      float v = acc[nt][r] + bv;
      if (relu) v = fmaxf(v, 0.f);
      outLds[row * 128 + col] = v;
    }
  }
}

// ---------------------------------------------------------------------------
// Utility kernels
// ---------------------------------------------------------------------------
// Convert f32 weights (K x 128, row-major) into WMMA-fragment-packed f16.
__global__ __launch_bounds__(256) void pack_w_f16(const float* __restrict__ s,
                                                  _Float16* __restrict__ d, int K) {
  int o = blockIdx.x * 256 + threadIdx.x;
  if (o >= K * 128) return;
  int h    = o & 15;
  int l    = (o >> 4) & 31;
  int frag = o >> 9;            // kt*8 + nt
  int nt   = frag & 7;
  int kt   = frag >> 3;
  int k    = kt * 32 + (h >> 3) * 16 + (l >> 4) * 8 + (h & 7);
  int col  = nt * 16 + (l & 15);
  d[o] = (_Float16)s[(size_t)k * 128 + col];
}

__global__ __launch_bounds__(256) void fill_f32(float* p, float v, size_t n) {
  size_t i = (size_t)blockIdx.x * 256 + threadIdx.x;
  if (i < n) p[i] = v;
}

// ---------------------------------------------------------------------------
// Embedder: nodes = concat(species_emb[species], LN(relu(pos @ W3x256 + b))) @ Wproj + bproj
// ---------------------------------------------------------------------------
__global__ __launch_bounds__(32) void embed_kernel(
    const float* __restrict__ positions, const int* __restrict__ species,
    const float* __restrict__ species_emb,
    const float* __restrict__ pmw, const float* __restrict__ pmb,
    const float* __restrict__ pmg, const float* __restrict__ pmbeta,
    const _Float16* __restrict__ npw, const float* __restrict__ npb,
    float* __restrict__ nodes, int N) {
  __shared__ float X[16 * 384];
  __shared__ float Z[16 * 128];
  const int n0 = blockIdx.x * 16;
  const int lane = threadIdx.x;
  __builtin_prefetch(npw);

  for (int i = lane; i < 16 * 32; i += 32) {
    int row = i >> 5, c4 = i & 31;
    int n = n0 + row; if (n >= N) n = N - 1;
    int sp = species[n];
    ((float4*)(X + row * 384))[c4] = ((const float4*)(species_emb + (size_t)sp * 128))[c4];
  }
  for (int i = lane; i < 16 * 256; i += 32) {
    int row = i >> 8, col = i & 255;
    int n = n0 + row; if (n >= N) n = N - 1;
    const float* p = positions + (size_t)n * 3;
    float v = pmb[col] + p[0] * pmw[col] + p[1] * pmw[256 + col] + p[2] * pmw[512 + col];
    X[row * 384 + 128 + col] = fmaxf(v, 0.f);
  }
  __syncthreads();
  ln_rows(X + 128, 384, 256, pmg, pmbeta, 16);
  __syncthreads();
  gemm_tile(X, 384, npw, npb, false, Z);   // node_proj (linear)
  __syncthreads();
  for (int i = lane; i < 2048; i += 32) {
    int row = i >> 7, col = i & 127;
    int n = n0 + row;
    if (n < N) nodes[(size_t)n * 128 + col] = Z[row * 128 + col];
  }
}

// ---------------------------------------------------------------------------
// Edge update
// ---------------------------------------------------------------------------
__global__ __launch_bounds__(32) void edge_step_kernel(
    float* __restrict__ edges, const float* __restrict__ nodes,
    const float* __restrict__ glob,
    const int* __restrict__ senders, const int* __restrict__ receivers,
    const _Float16* __restrict__ w0, const float* __restrict__ b0,
    const float* __restrict__ g0, const float* __restrict__ be0,
    const _Float16* __restrict__ w1, const float* __restrict__ b1,
    const float* __restrict__ g1, const float* __restrict__ be1,
    const float* __restrict__ lng, const float* __restrict__ lnbeta,
    float* __restrict__ sent, float* __restrict__ recv,
    float* __restrict__ edge_agg, int E, int EperG) {
  __shared__ float X[16 * 512];
  __shared__ float Y[16 * 128];
  __shared__ float Z[16 * 128];
  const int e0 = blockIdx.x * 16;
  const int lane = threadIdx.x;
  __builtin_prefetch(w0);
  __builtin_prefetch(w1);

  for (int i = lane; i < 2048; i += 32) {
    int row = i >> 7, c4 = i & 127;
    int seg = c4 >> 5, col = c4 & 31;
    int e = e0 + row; if (e >= E) e = E - 1;
    const float* src;
    if (seg == 0)      src = edges + (size_t)e * 128;
    else if (seg == 1) src = nodes + (size_t)senders[e] * 128;
    else if (seg == 2) src = nodes + (size_t)receivers[e] * 128;
    else               src = glob  + (size_t)(e / EperG) * 128;
    ((float4*)(X + row * 512 + seg * 128))[col] = ((const float4*)src)[col];
  }
  __syncthreads();
  gemm_tile(X, 512, w0, b0, true, Y);
  __syncthreads();
  ln_rows(Y, 128, 128, g0, be0, 16);
  __syncthreads();
  gemm_tile(Y, 128, w1, b1, true, Z);
  __syncthreads();
  ln_rows(Z, 128, 128, g1, be1, 16);     // Z = new_edges
  __syncthreads();

  for (int i = lane; i < 2048; i += 32) {
    int row = i >> 7, col = i & 127;
    int e = e0 + row;
    if (e < E) {
      float v = Z[row * 128 + col];
      atomicAdd(sent + (size_t)senders[e] * 128 + col, v);
      atomicAdd(recv + (size_t)receivers[e] * 128 + col, v);
      atomicAdd(edge_agg + (size_t)(e / EperG) * 128 + col, v);
    }
  }
  __syncthreads();
  for (int i = lane; i < 2048; i += 32) {
    int row = i >> 7, col = i & 127;
    Z[row * 128 + col] += X[row * 512 + col];   // old edges
  }
  __syncthreads();
  ln_rows(Z, 128, 128, lng, lnbeta, 16);
  __syncthreads();
  for (int i = lane; i < 2048; i += 32) {
    int row = i >> 7, col = i & 127;
    int e = e0 + row;
    if (e < E) edges[(size_t)e * 128 + col] = Z[row * 128 + col];
  }
}

// ---------------------------------------------------------------------------
// Node update
// ---------------------------------------------------------------------------
__global__ __launch_bounds__(32) void node_step_kernel(
    const float* __restrict__ nodes, const float* __restrict__ sent,
    const float* __restrict__ recv, const float* __restrict__ glob,
    const _Float16* __restrict__ w0, const float* __restrict__ b0,
    const float* __restrict__ g0, const float* __restrict__ be0,
    const _Float16* __restrict__ w1, const float* __restrict__ b1,
    const float* __restrict__ g1, const float* __restrict__ be1,
    const float* __restrict__ lng, const float* __restrict__ lnbeta,
    float* __restrict__ nodes_out, float* __restrict__ node_agg,
    int N, int NperG) {
  __shared__ float X[16 * 512];
  __shared__ float Y[16 * 128];
  __shared__ float Z[16 * 128];
  const int n0 = blockIdx.x * 16;
  const int lane = threadIdx.x;
  __builtin_prefetch(w0);
  __builtin_prefetch(w1);

  for (int i = lane; i < 2048; i += 32) {
    int row = i >> 7, c4 = i & 127;
    int seg = c4 >> 5, col = c4 & 31;
    int n = n0 + row; if (n >= N) n = N - 1;
    const float* src;
    if (seg == 0)      src = nodes + (size_t)n * 128;
    else if (seg == 1) src = sent  + (size_t)n * 128;
    else if (seg == 2) src = recv  + (size_t)n * 128;
    else               src = glob  + (size_t)(n / NperG) * 128;
    ((float4*)(X + row * 512 + seg * 128))[col] = ((const float4*)src)[col];
  }
  __syncthreads();
  gemm_tile(X, 512, w0, b0, true, Y);
  __syncthreads();
  ln_rows(Y, 128, 128, g0, be0, 16);
  __syncthreads();
  gemm_tile(Y, 128, w1, b1, true, Z);
  __syncthreads();
  ln_rows(Z, 128, 128, g1, be1, 16);     // Z = new_nodes
  __syncthreads();

  for (int i = lane; i < 2048; i += 32) {
    int row = i >> 7, col = i & 127;
    int n = n0 + row;
    if (n < N)
      atomicAdd(node_agg + (size_t)(n / NperG) * 128 + col, Z[row * 128 + col]);
  }
  __syncthreads();
  for (int i = lane; i < 2048; i += 32) {
    int row = i >> 7, col = i & 127;
    Z[row * 128 + col] += X[row * 512 + col];   // old nodes
  }
  __syncthreads();
  ln_rows(Z, 128, 128, lng, lnbeta, 16);
  __syncthreads();
  for (int i = lane; i < 2048; i += 32) {
    int row = i >> 7, col = i & 127;
    int n = n0 + row;
    if (n < N) nodes_out[(size_t)n * 128 + col] = Z[row * 128 + col];
  }
}

// ---------------------------------------------------------------------------
// Global update
// ---------------------------------------------------------------------------
__global__ __launch_bounds__(32) void global_step_kernel(
    const float* __restrict__ node_agg, const float* __restrict__ edge_agg,
    const float* __restrict__ glob,
    const _Float16* __restrict__ w0, const float* __restrict__ b0,
    const float* __restrict__ g0, const float* __restrict__ be0,
    const _Float16* __restrict__ w1, const float* __restrict__ b1,
    const float* __restrict__ g1, const float* __restrict__ be1,
    const float* __restrict__ lng, const float* __restrict__ lnbeta,
    float* __restrict__ glob_out, int G) {
  __shared__ float X[16 * 384];
  __shared__ float Y[16 * 128];
  __shared__ float Z[16 * 128];
  const int g0r = blockIdx.x * 16;
  const int lane = threadIdx.x;
  int nvalid = G - g0r; if (nvalid > 16) nvalid = 16; if (nvalid < 0) nvalid = 0;

  for (int i = lane; i < 16 * 96; i += 32) {
    int row = i / 96, c4 = i % 96;
    int seg = c4 / 32, col = c4 % 32;
    int g = g0r + row;
    float4 v = make_float4(0.f, 0.f, 0.f, 0.f);
    if (g < G) {
      const float* src = (seg == 0) ? node_agg + (size_t)g * 128
                        : (seg == 1) ? edge_agg + (size_t)g * 128
                                     : glob + (size_t)g * 128;
      v = ((const float4*)src)[col];
    }
    ((float4*)(X + row * 384 + seg * 128))[col] = v;
  }
  __syncthreads();
  gemm_tile(X, 384, w0, b0, true, Y);
  __syncthreads();
  ln_rows(Y, 128, 128, g0, be0, nvalid);
  __syncthreads();
  gemm_tile(Y, 128, w1, b1, true, Z);
  __syncthreads();
  ln_rows(Z, 128, 128, g1, be1, nvalid);
  __syncthreads();
  for (int i = lane; i < 2048; i += 32) {
    int row = i >> 7, col = i & 127;
    Z[row * 128 + col] += X[row * 384 + 256 + col];   // old glob
  }
  __syncthreads();
  ln_rows(Z, 128, 128, lng, lnbeta, nvalid);
  __syncthreads();
  for (int i = lane; i < 2048; i += 32) {
    int row = i >> 7, col = i & 127;
    int g = g0r + row;
    if (g < G) glob_out[(size_t)g * 128 + col] = Z[row * 128 + col];
  }
}

// ---------------------------------------------------------------------------
// Heads
// ---------------------------------------------------------------------------
__global__ __launch_bounds__(256) void focus_kernel(
    const float* __restrict__ nodes, const float* __restrict__ fw,
    const float* __restrict__ fb, float* __restrict__ out, int N) {
  int n = blockIdx.x * 256 + threadIdx.x;
  if (n >= N) return;
  const float* r = nodes + (size_t)n * 128;
  float s = fb[0];
  for (int i = 0; i < 128; ++i) s += r[i] * fw[i];
  out[n] = s;
}

__global__ __launch_bounds__(256) void graph_head_kernel(
    const float* __restrict__ nodes, const float* __restrict__ species_emb,
    const int* __restrict__ target_species,
    const float* __restrict__ shw, const float* __restrict__ shb,
    const float* __restrict__ phw, const float* __restrict__ phb,
    float* __restrict__ species_out, float* __restrict__ pos_out,
    int NperG, int NUM_ELEM) {
  __shared__ float pin[256];
  const int g = blockIdx.x;
  const int t = threadIdx.x;
  if (t < 128) pin[t] = nodes[(size_t)g * NperG * 128 + t];
  else pin[t] = species_emb[(size_t)target_species[g] * 128 + (t - 128)];
  __syncthreads();
  if (t < NUM_ELEM) {
    float s = shb[t];
    for (int k = 0; k < 128; ++k) s += pin[k] * shw[k * NUM_ELEM + t];
    species_out[g * NUM_ELEM + t] = s;
  }
  for (int j = t; j < 2304; j += 256) {
    float s = phb[j];
    for (int k = 0; k < 256; ++k) s += pin[k] * phw[(size_t)k * 2304 + j];
    pos_out[(size_t)g * 2304 + j] = s;
  }
}

// ---------------------------------------------------------------------------
// Host driver.  Input leaf order = jax tree_flatten (dict keys sorted).
// ---------------------------------------------------------------------------
extern "C" void kernel_launch(void* const* d_in, const int* in_sizes, int n_in,
                              void* d_out, int out_size, void* d_ws, size_t ws_size,
                              hipStream_t stream) {
  (void)n_in; (void)out_size; (void)ws_size;
  enum {
    I_FOCUS_B = 2, I_FOCUS_W = 3, I_NP_B = 4, I_NP_W = 5,
    I_PH_B = 6, I_PH_W = 7,
    I_PM_B = 8, I_PM_BETA = 9, I_PM_G = 10, I_PM_W = 11,
    I_SPEMB = 12, I_SH_B = 13, I_SH_W = 14,
    I_STEP0 = 15, STEP_STRIDE = 30,
    I_POS = 135, I_RECV = 136, I_SEND = 137, I_SPEC = 138, I_TGT = 139
  };
  const int G = in_sizes[0];              // n_edge has G entries
  const int E = in_sizes[I_SEND];
  const int N = in_sizes[I_SPEC];
  const int NperG = N / G;
  const int EperG = E / G;
  const int NUM_ELEM = in_sizes[I_SH_B];  // 5

  auto F = [&](int i) { return (const float*)d_in[i]; };

  // ---- workspace carve ----
  size_t off = 0;
  auto carve = [&](size_t bytes) -> void* {
    void* p = (char*)d_ws + off;
    off += (bytes + 255) & ~(size_t)255;
    return p;
  };
  float* nodesA   = (float*)carve((size_t)N * 128 * 4);
  float* nodesB   = (float*)carve((size_t)N * 128 * 4);
  float* sent     = (float*)carve((size_t)N * 128 * 4);
  float* recv     = (float*)carve((size_t)N * 128 * 4);
  float* globA    = (float*)carve((size_t)G * 128 * 4);
  float* globB    = (float*)carve((size_t)G * 128 * 4);
  float* node_agg = (float*)carve((size_t)G * 128 * 4);
  float* edge_agg = (float*)carve((size_t)G * 128 * 4);
  float* edges    = (float*)carve((size_t)E * 128 * 4);
  _Float16* h_np  = (_Float16*)carve((size_t)384 * 128 * 2);
  _Float16 *h_e0[4], *h_e1[4], *h_n0[4], *h_n1[4], *h_g0[4], *h_g1[4];
  for (int s = 0; s < 4; ++s) {
    h_e0[s] = (_Float16*)carve((size_t)512 * 128 * 2);
    h_e1[s] = (_Float16*)carve((size_t)128 * 128 * 2);
    h_n0[s] = (_Float16*)carve((size_t)512 * 128 * 2);
    h_n1[s] = (_Float16*)carve((size_t)128 * 128 * 2);
    h_g0[s] = (_Float16*)carve((size_t)384 * 128 * 2);
    h_g1[s] = (_Float16*)carve((size_t)128 * 128 * 2);
  }

  // ---- pack all WMMA weights into fragment-ordered f16 (L2-resident) ----
  auto pack = [&](int idx, _Float16* dst, int K) {
    int n = K * 128;
    pack_w_f16<<<(n + 255) / 256, 256, 0, stream>>>(F(idx), dst, K);
  };
  pack(I_NP_W, h_np, 384);
  for (int s = 0; s < 4; ++s) {
    const int b = I_STEP0 + STEP_STRIDE * s;
    pack(b + 3,  h_e0[s], 512);
    pack(b + 7,  h_e1[s], 128);
    pack(b + 11, h_g0[s], 384);
    pack(b + 15, h_g1[s], 128);
    pack(b + 25, h_n0[s], 512);
    pack(b + 29, h_n1[s], 128);
  }

  // ---- initial features: edges = 1, glob = 1 ----
  {
    size_t ne = (size_t)E * 128;
    fill_f32<<<(unsigned)((ne + 255) / 256), 256, 0, stream>>>(edges, 1.0f, ne);
    size_t ng = (size_t)G * 128;
    fill_f32<<<(unsigned)((ng + 255) / 256), 256, 0, stream>>>(globA, 1.0f, ng);
  }

  // ---- embedder ----
  embed_kernel<<<(N + 15) / 16, 32, 0, stream>>>(
      F(I_POS), (const int*)d_in[I_SPEC], F(I_SPEMB),
      F(I_PM_W), F(I_PM_B), F(I_PM_G), F(I_PM_BETA),
      h_np, F(I_NP_B), nodesA, N);

  // ---- message passing steps ----
  float *ncur = nodesA, *nnext = nodesB, *gcur = globA, *gnext = globB;
  const int* senders   = (const int*)d_in[I_SEND];
  const int* receivers = (const int*)d_in[I_RECV];
  for (int s = 0; s < 4; ++s) {
    const int b = I_STEP0 + STEP_STRIDE * s;
    hipMemsetAsync(sent, 0, (size_t)N * 128 * 4, stream);
    hipMemsetAsync(recv, 0, (size_t)N * 128 * 4, stream);
    hipMemsetAsync(node_agg, 0, (size_t)G * 128 * 4, stream);
    hipMemsetAsync(edge_agg, 0, (size_t)G * 128 * 4, stream);

    edge_step_kernel<<<(E + 15) / 16, 32, 0, stream>>>(
        edges, ncur, gcur, senders, receivers,
        h_e0[s], F(b + 0), F(b + 2), F(b + 1),
        h_e1[s], F(b + 4), F(b + 6), F(b + 5),
        F(b + 17), F(b + 16),              // ln_edges: g, beta
        sent, recv, edge_agg, E, EperG);

    node_step_kernel<<<(N + 15) / 16, 32, 0, stream>>>(
        ncur, sent, recv, gcur,
        h_n0[s], F(b + 22), F(b + 24), F(b + 23),
        h_n1[s], F(b + 26), F(b + 28), F(b + 27),
        F(b + 21), F(b + 20),              // ln_nodes: g, beta
        nnext, node_agg, N, NperG);

    global_step_kernel<<<(G + 15) / 16, 32, 0, stream>>>(
        node_agg, edge_agg, gcur,
        h_g0[s], F(b + 8), F(b + 10), F(b + 9),
        h_g1[s], F(b + 12), F(b + 14), F(b + 13),
        F(b + 19), F(b + 18),              // ln_globals: g, beta
        gnext, G);

    { float* t = ncur; ncur = nnext; nnext = t; }
    { float* t = gcur; gcur = gnext; gnext = t; }
  }

  // ---- heads ----
  float* focus_out   = (float*)d_out;
  float* species_out = focus_out + N;
  float* pos_out     = species_out + (size_t)G * NUM_ELEM;
  focus_kernel<<<(N + 255) / 256, 256, 0, stream>>>(ncur, F(I_FOCUS_W), F(I_FOCUS_B),
                                                    focus_out, N);
  graph_head_kernel<<<G, 256, 0, stream>>>(
      ncur, F(I_SPEMB), (const int*)d_in[I_TGT],
      F(I_SH_W), F(I_SH_B), F(I_PH_W), F(I_PH_B),
      species_out, pos_out, NperG, NUM_ELEM);
}
